// FlaxGemma2DecoderLayer_23407571763832
// MI455X (gfx1250) — matrix operand order repaired and checked by reading, not compiled, not run on hardware
//
#include <hip/hip_runtime.h>
#include <hip/hip_bf16.h>

// ---------------------------------------------------------------------------
// CDNA5 (gfx1250) Gemma2 decoder layer. wave32, WMMA bf16 (fp32 accumulate).
// ---------------------------------------------------------------------------

typedef __bf16 v16bf __attribute__((ext_vector_type(16)));
typedef float  v8f   __attribute__((ext_vector_type(8)));

union FragBF { v16bf v; unsigned short u16[16]; unsigned int u32[8]; };
union AccF   { v8f v; float f[8]; };

// Native fptrunc -> backend picks hardware bf16 cvt on gfx1250 (RNE).
__device__ __forceinline__ unsigned short f2bf(float f) {
    return __builtin_bit_cast(unsigned short, (__bf16)f);
}
__device__ __forceinline__ unsigned pack_bf16(float lo, float hi) {
    union { __bf16 b[2]; unsigned u; } pk;
    pk.b[0] = (__bf16)lo;
    pk.b[1] = (__bf16)hi;
    return pk.u;
}

// ---------------------------------------------------------------------------
// RMSNorm: out = (resid?) + (1+w) * x * rsqrt(mean(x^2)+eps).  One row / WG.
// ---------------------------------------------------------------------------
__global__ __launch_bounds__(256) void rmsnorm_kernel(
    const float* __restrict__ x, const float* __restrict__ w,
    const float* __restrict__ resid, float* __restrict__ out, int D)
{
    __shared__ float red[8];
    const int row = blockIdx.x;
    const float* xr = x + (long)row * D;
    float s = 0.f;
    for (int i = threadIdx.x; i < D; i += 256) { float v = xr[i]; s += v * v; }
#pragma unroll
    for (int m = 16; m >= 1; m >>= 1) s += __shfl_xor(s, m, 32);
    if ((threadIdx.x & 31) == 0) red[threadIdx.x >> 5] = s;
    __syncthreads();
    if (threadIdx.x < 8) {
        float t = red[threadIdx.x];
#pragma unroll
        for (int m = 4; m >= 1; m >>= 1) t += __shfl_xor(t, m, 32);
        if (threadIdx.x == 0) red[0] = t;
    }
    __syncthreads();
    const float rstd = rsqrtf(red[0] / (float)D + 1e-6f);
    for (int i = threadIdx.x; i < D; i += 256) {
        float v = (1.0f + w[i]) * (xr[i] * rstd);
        if (resid) v += resid[(long)row * D + i];
        out[(long)row * D + i] = v;
    }
}

// ---------------------------------------------------------------------------
// Tiled WMMA GEMM: C[M,N] = A[M,K] * B[K,N], fp32 in, bf16 matrix-core compute,
// fp32 accumulate/out. Block tile 64x128x32, 8 waves, wave tile 32x32.
// mode 0: C = acc.  mode 2: C = gelu_tanh(aux[idx]) * acc  (fused gelu(gate)*up)
// ---------------------------------------------------------------------------
__global__ __launch_bounds__(256) void gemm_bf16_wmma(
    const float* __restrict__ A, const float* __restrict__ B,
    float* __restrict__ C, const float* __restrict__ aux,
    int M, int N, int K, int mode)
{
    __shared__ unsigned int As32[64 * 16];    // [m][k] bf16 pairs (k fastest)
    __shared__ unsigned int Bs32[128 * 16];   // [n][k] bf16 pairs (transposed)

    const int tid  = threadIdx.x;
    const int lane = tid & 31;
    const int wave = tid >> 5;
    const int wm   = wave >> 2;          // 0..1
    const int wn   = wave & 3;           // 0..3
    const int half = lane >> 4;
    const int l16  = lane & 15;
    const int bm0  = blockIdx.y * 64;
    const int bn0  = blockIdx.x * 128;

    AccF acc[2][2];
#pragma unroll
    for (int i = 0; i < 2; ++i)
#pragma unroll
        for (int j = 0; j < 2; ++j)
#pragma unroll
            for (int e = 0; e < 8; ++e) acc[i][j].f[e] = 0.f;

    for (int k0 = 0; k0 < K; k0 += 32) {
        // stage A tile 64x32: float2 global loads, packed b32 LDS stores
#pragma unroll
        for (int e = 0; e < 4; ++e) {
            int idx = tid + e * 256;             // uint index, 1024 total
            int r = idx >> 4, c = (idx & 15) * 2;
            float2 f = *(const float2*)&A[(long)(bm0 + r) * K + k0 + c];
            As32[idx] = pack_bf16(f.x, f.y);
        }
        // stage B tile 32x128 -> [n][k] transposed; coalesced over n
#pragma unroll
        for (int e = 0; e < 8; ++e) {
            int idx = tid + e * 256;             // 2048 total
            int c  = idx & 127;                  // n
            int rp = idx >> 7;                   // k pair 0..15
            const float* bp = &B[(long)(k0 + rp * 2) * N + bn0 + c];
            Bs32[c * 16 + rp] = pack_bf16(bp[0], bp[N]);
        }
        // prefetch next K tile while this one is consumed
        if (k0 + 32 < K) {
            __builtin_prefetch(&A[(long)(bm0 + (tid & 63)) * K + k0 + 32], 0, 1);
            __builtin_prefetch(&B[(long)(k0 + 32 + (tid >> 3)) * N + bn0 + (tid & 7) * 16], 0, 1);
        }
        __syncthreads();

        FragBF a[2], b[2];
#pragma unroll
        for (int mt = 0; mt < 2; ++mt) {
            int row = wm * 32 + mt * 16 + l16;          // A: lane holds row M
#pragma unroll
            for (int j = 0; j < 8; ++j) {               // k = half*16 + i contiguous
                int off = half * 8 + j;
                a[mt].u32[j] = As32[row * 16 + off];
            }
        }
#pragma unroll
        for (int nt = 0; nt < 2; ++nt) {
            int nn = wn * 32 + nt * 16 + l16;           // B: lane holds col N
#pragma unroll
            for (int j = 0; j < 8; ++j)                 // k = half*16 + i contiguous
                b[nt].u32[j] = Bs32[nn * 16 + half * 8 + j];
        }
#pragma unroll
        for (int mt = 0; mt < 2; ++mt)
#pragma unroll
            for (int nt = 0; nt < 2; ++nt)
                acc[mt][nt].v = __builtin_amdgcn_wmma_f32_16x16x32_bf16(
                    false, a[mt].v, false, b[nt].v, (short)0, acc[mt][nt].v,
                    false, false);
        __syncthreads();
    }

    // epilogue: C/D layout -> row = r + 8*half, col = l16
#pragma unroll
    for (int mt = 0; mt < 2; ++mt)
#pragma unroll
        for (int nt = 0; nt < 2; ++nt)
#pragma unroll
            for (int r = 0; r < 8; ++r) {
                int row = bm0 + wm * 32 + mt * 16 + r + 8 * half;
                int col = bn0 + wn * 32 + nt * 16 + l16;
                long o = (long)row * N + col;
                float v = acc[mt][nt].f[r];
                if (mode == 2) {
                    // gelu_tanh(g)*v == g * sigmoid(2*t) * v, branch-free
                    float g = aux[o];
                    float t2 = 1.5957691216057308f * (g + 0.044715f * g * g * g);
                    float e = __expf(t2);
                    v = g * (e / (e + 1.0f)) * v;
                }
                C[o] = v;
            }
}

// ---------------------------------------------------------------------------
// RoPE (rotate-half over full head_dim=256) + fp32->bf16, relayout to [h][S][256]
// ---------------------------------------------------------------------------
__global__ void rope_kernel(const float* __restrict__ lin, const int* __restrict__ pos,
                            unsigned short* __restrict__ outbf, int nheads)
{
    int idx = blockIdx.x * blockDim.x + threadIdx.x;   // pair index
    int d = idx & 127;
    int h = (idx >> 7) % nheads;
    int s = idx / (128 * nheads);
    const float* p = lin + (long)s * nheads * 256 + h * 256;
    float x0 = p[d], x1 = p[d + 128];
    float f = (float)pos[s] * __powf(10000.0f, -(float)d * (1.0f / 128.0f));
    float sn, c;
    __sincosf(f, &sn, &c);
    unsigned short* o = outbf + ((long)h * 2048 + s) * 256;
    o[d]       = f2bf(x0 * c - x1 * sn);
    o[d + 128] = f2bf(x1 * c + x0 * sn);
}

__global__ void vconv_kernel(const float* __restrict__ lin,
                             unsigned short* __restrict__ outbf, int nheads)
{
    int idx = blockIdx.x * blockDim.x + threadIdx.x;
    int d = idx & 255;
    int h = (idx >> 8) % nheads;
    int s = idx / (256 * nheads);
    outbf[((long)h * 2048 + s) * 256 + d] = f2bf(lin[(long)s * nheads * 256 + h * 256 + d]);
}

// ---------------------------------------------------------------------------
// Flash attention, sliding window 1024, GQA 2:1, all matmuls via WMMA bf16.
// WG = (16-query tile, head); 8 waves, wave w owns output dims [32w, 32w+32).
// K fragments read bf16 global directly; V tile staged to LDS transposed.
// ---------------------------------------------------------------------------
__global__ __launch_bounds__(256) void attn_kernel(
    const unsigned short* __restrict__ qbf,   // [8][2048][256] bf16
    const unsigned short* __restrict__ kbf,   // [4][2048][256] bf16
    const unsigned short* __restrict__ vbf,   // [4][2048][256] bf16
    const int* __restrict__ amask,            // [2048]
    float* __restrict__ out)                  // [2048][2048] fp32
{
    const int S = 2048, HD = 256, SW = 1024;
    const float SCALE = 0.0625f;              // 256^-0.5
    __shared__ unsigned int   Qs[16 * 128];   // 16 x 256 bf16 (uint pairs)   8KB
    __shared__ unsigned short Ps[8][16 * 32]; // per-wave P tile (bf16)       8KB
    __shared__ unsigned short Vs[256 * 32];   // V tile transposed [dim][key] 16KB

    const int q0   = blockIdx.x * 16;
    const int head = blockIdx.y;
    const int kvh  = head >> 1;
    const int tid  = threadIdx.x;
    const int lane = tid & 31;
    const int wave = tid >> 5;
    const int half = lane >> 4;
    const int l16  = lane & 15;

    const unsigned int* qsrc = (const unsigned int*)(qbf + ((long)head * S + q0) * HD);
#pragma unroll
    for (int e = 0; e < 8; ++e) Qs[tid + e * 256] = qsrc[tid + e * 256];
    __syncthreads();

    AccF o[2];
#pragma unroll
    for (int nt = 0; nt < 2; ++nt)
#pragma unroll
        for (int r = 0; r < 8; ++r) o[nt].f[r] = 0.f;
    float mrow[8], lrow[8];
#pragma unroll
    for (int r = 0; r < 8; ++r) { mrow[r] = -1e30f; lrow[r] = 0.f; }

    const unsigned int* kp   = (const unsigned int*)kbf;
    const unsigned int* vp32 = (const unsigned int*)vbf;
    int jlo = q0 - SW + 1; if (jlo < 0) jlo = 0; jlo &= ~31;

    for (int j0 = jlo; j0 <= q0 + 15; j0 += 32) {
        // ---- scores S = Q * K^T  (16 x 32 keys), K-dim 256 in 8 chunks ----
        AccF sc[2];
#pragma unroll
        for (int nt = 0; nt < 2; ++nt)
#pragma unroll
            for (int r = 0; r < 8; ++r) sc[nt].f[r] = 0.f;
#pragma unroll
        for (int kk = 0; kk < 8; ++kk) {
            FragBF a;
#pragma unroll
            for (int j = 0; j < 8; ++j) {
                int off = half * 8 + j;
                a.u32[j] = Qs[l16 * 128 + kk * 16 + off];
            }
#pragma unroll
            for (int nt = 0; nt < 2; ++nt) {
                int key = j0 + nt * 16 + l16;
                FragBF b;
                long base = ((long)kvh * S + key) * 128 + kk * 16 + half * 8;
#pragma unroll
                for (int j = 0; j < 8; ++j) b.u32[j] = kp[base + j];
                sc[nt].v = __builtin_amdgcn_wmma_f32_16x16x32_bf16(
                    false, a.v, false, b.v, (short)0, sc[nt].v, false, false);
            }
        }
        // ---- scale + causal/window/padding mask ----
#pragma unroll
        for (int nt = 0; nt < 2; ++nt) {
            int jj = j0 + nt * 16 + l16;
            bool mvalid = amask[jj] > 0;
#pragma unroll
            for (int r = 0; r < 8; ++r) {
                int ii = q0 + r + 8 * half;
                bool keep = mvalid && (jj <= ii) && ((ii - jj) < SW);
                sc[nt].f[r] = keep ? sc[nt].f[r] * SCALE : -1e30f;
            }
        }
        // ---- online softmax (row reductions across 16-lane groups) ----
        float pr[2][8];
#pragma unroll
        for (int r = 0; r < 8; ++r) {
            float mx = fmaxf(sc[0].f[r], sc[1].f[r]);
#pragma unroll
            for (int m = 8; m >= 1; m >>= 1) mx = fmaxf(mx, __shfl_xor(mx, m, 32));
            float mn  = fmaxf(mrow[r], mx);
            float fac = __expf(mrow[r] - mn);
            float p0  = __expf(sc[0].f[r] - mn);
            float p1  = __expf(sc[1].f[r] - mn);
            float rs  = p0 + p1;
#pragma unroll
            for (int m = 8; m >= 1; m >>= 1) rs += __shfl_xor(rs, m, 32);
            lrow[r] = lrow[r] * fac + rs;
            mrow[r] = mn;
            o[0].f[r] *= fac;
            o[1].f[r] *= fac;
            pr[0][r] = p0; pr[1][r] = p1;
        }
        // ---- stage P (C-layout -> A-layout) and V tile (transposed) ----
        __syncthreads();                       // protect vs prev iter's reads
#pragma unroll
        for (int nt = 0; nt < 2; ++nt)
#pragma unroll
            for (int r = 0; r < 8; ++r)
                Ps[wave][(r + 8 * half) * 32 + nt * 16 + l16] = f2bf(pr[nt][r]);
#pragma unroll
        for (int e = 0; e < 16; ++e) {
            int idx = tid + e * 256;           // 4096 uints: 32 keys x 128 pairs
            int k  = idx >> 7;                 // key 0..31
            int dp = idx & 127;                // dim pair
            unsigned v = vp32[((long)kvh * S + j0 + k) * 128 + dp];
            Vs[(2 * dp) * 32 + k]     = (unsigned short)(v & 0xFFFFu);
            Vs[(2 * dp + 1) * 32 + k] = (unsigned short)(v >> 16);
        }
        __syncthreads();

        FragBF pa;
        const unsigned int* ps32 = (const unsigned int*)Ps[wave];
#pragma unroll
        for (int j = 0; j < 8; ++j) {
            int off = half * 8 + j;
            pa.u32[j] = ps32[l16 * 16 + off];
        }
        // ---- O += P * V  (wave's 32 dims = 2 n-tiles) ----
        const unsigned int* vs32 = (const unsigned int*)Vs;
#pragma unroll
        for (int nt = 0; nt < 2; ++nt) {
            FragBF vb;
            int n = wave * 32 + nt * 16 + l16;
#pragma unroll
            for (int j = 0; j < 8; ++j)
                vb.u32[j] = vs32[n * 16 + half * 8 + j];
            o[nt].v = __builtin_amdgcn_wmma_f32_16x16x32_bf16(
                false, pa.v, false, vb.v, (short)0, o[nt].v, false, false);
        }
    }
    // ---- finalize: O / l ----
#pragma unroll
    for (int r = 0; r < 8; ++r) {
        float inv = 1.0f / lrow[r];
        int row = q0 + r + 8 * half;
#pragma unroll
        for (int nt = 0; nt < 2; ++nt) {
            int col = head * HD + wave * 32 + nt * 16 + l16;
            out[(long)row * 2048 + col] = o[nt].f[r] * inv;
        }
    }
}

// ---------------------------------------------------------------------------
extern "C" void kernel_launch(void* const* d_in, const int* in_sizes, int n_in,
                              void* d_out, int out_size, void* d_ws, size_t ws_size,
                              hipStream_t stream)
{
    (void)in_sizes; (void)n_in; (void)out_size; (void)ws_size;
    const float* hidden = (const float*)d_in[0];
    const int*   amask  = (const int*)d_in[1];
    const int*   pos    = (const int*)d_in[2];
    const float* wq     = (const float*)d_in[3];
    const float* wk     = (const float*)d_in[4];
    const float* wv     = (const float*)d_in[5];
    const float* wo     = (const float*)d_in[6];
    const float* wg     = (const float*)d_in[7];
    const float* wu     = (const float*)d_in[8];
    const float* wd     = (const float*)d_in[9];
    const float* ln_in  = (const float*)d_in[10];
    const float* ln_pa  = (const float*)d_in[11];
    const float* ln_pf  = (const float*)d_in[12];
    const float* ln_po  = (const float*)d_in[13];
    float* out = (float*)d_out;

    const int s_ = 2048, d_ = 2304, i_ = 9216;
    char* p = (char*)d_ws;
    auto carve = [&](size_t bytes) { void* r = (void*)p; p += (bytes + 255) & ~(size_t)255; return r; };
    float* h1   = (float*)carve((size_t)s_ * d_ * 4);
    float* qlin = (float*)carve((size_t)s_ * 2048 * 4);
    float* klin = (float*)carve((size_t)s_ * 1024 * 4);
    float* vlin = (float*)carve((size_t)s_ * 1024 * 4);
    unsigned short* qb = (unsigned short*)carve((size_t)s_ * 2048 * 2);
    unsigned short* kb = (unsigned short*)carve((size_t)s_ * 1024 * 2);
    unsigned short* vb = (unsigned short*)carve((size_t)s_ * 1024 * 2);
    float* attn = (float*)carve((size_t)s_ * 2048 * 4);
    float* ao   = (float*)carve((size_t)s_ * d_ * 4);
    float* x    = (float*)carve((size_t)s_ * d_ * 4);
    float* h2   = (float*)carve((size_t)s_ * d_ * 4);
    float* gate = (float*)carve((size_t)s_ * i_ * 4);
    float* h3   = (float*)carve((size_t)s_ * i_ * 4);
    float* mlp  = (float*)carve((size_t)s_ * d_ * 4);

    dim3 b256(256);
    // 1) input norm
    rmsnorm_kernel<<<s_, b256, 0, stream>>>(hidden, ln_in, nullptr, h1, d_);
    // 2) QKV projections
    gemm_bf16_wmma<<<dim3(2048 / 128, s_ / 64), b256, 0, stream>>>(h1, wq, qlin, nullptr, s_, 2048, d_, 0);
    gemm_bf16_wmma<<<dim3(1024 / 128, s_ / 64), b256, 0, stream>>>(h1, wk, klin, nullptr, s_, 1024, d_, 0);
    gemm_bf16_wmma<<<dim3(1024 / 128, s_ / 64), b256, 0, stream>>>(h1, wv, vlin, nullptr, s_, 1024, d_, 0);
    // 3) RoPE + bf16 relayout
    rope_kernel<<<(s_ * 8 * 128) / 256, b256, 0, stream>>>(qlin, pos, qb, 8);
    rope_kernel<<<(s_ * 4 * 128) / 256, b256, 0, stream>>>(klin, pos, kb, 4);
    vconv_kernel<<<(s_ * 4 * 256) / 256, b256, 0, stream>>>(vlin, vb, 4);
    // 4) sliding-window flash attention
    attn_kernel<<<dim3(s_ / 16, 8), b256, 0, stream>>>(qb, kb, vb, amask, attn);
    // 5) output projection + sandwich norms
    gemm_bf16_wmma<<<dim3(d_ / 128, s_ / 64), b256, 0, stream>>>(attn, wo, ao, nullptr, s_, d_, 2048, 0);
    rmsnorm_kernel<<<s_, b256, 0, stream>>>(ao, ln_pa, hidden, x, d_);
    rmsnorm_kernel<<<s_, b256, 0, stream>>>(x, ln_pf, nullptr, h2, d_);
    // 6) MLP: gate, up (fused gelu(gate)*up), down
    gemm_bf16_wmma<<<dim3(i_ / 128, s_ / 64), b256, 0, stream>>>(h2, wg, gate, nullptr, s_, i_, d_, 0);
    gemm_bf16_wmma<<<dim3(i_ / 128, s_ / 64), b256, 0, stream>>>(h2, wu, h3, gate, s_, i_, d_, 2);
    gemm_bf16_wmma<<<dim3(d_ / 128, s_ / 64), b256, 0, stream>>>(h3, wd, mlp, nullptr, s_, d_, i_, 0);
    rmsnorm_kernel<<<s_, b256, 0, stream>>>(mlp, ln_po, x, out, d_);
}
